// BidLSTM_60327110640347
// MI455X (gfx1250) — matrix operand order repaired and checked by reading, not compiled, not run on hardware
//
#include <hip/hip_runtime.h>
#include <hip/hip_bf16.h>
#include <math.h>

// ---------------------------------------------------------------------------
// Bidirectional masked LSTM on MI455X (gfx1250), wave32 + WMMA f16->f32,
// with GLOBAL_LOAD_ASYNC_TO_LDS_B128 (ASYNCcnt) staging.
//
// Pass 0a: x  (fp32) -> Xh (f16)                  [40.96 MB, one-time]
// Pass 0b: W,U (fp32 row-major) -> f16 col-major  [3.28 MB,  one-time]
// Phase 1: G_d[t][b][1280] = f16( x @ W_d + b_d )  -- 64x64 tile per WG,
//          K-panels (32) double-buffered in LDS via async b128 copies,
//          V_WMMA_F32_16X16X32_F16.
// Phase 2: persistent recurrent scan, 20 WGs/direction, grid barrier per
//          step; each WG's transposed U-slice async-DMA'd into LDS once and
//          resident for all 1000 steps; c-state in LDS; h double-buffered
//          in global f16 (L2-resident).
//
// Workspace layout (needs ~372 MB), all offsets 16B-aligned:
//   [0)            Gf  : 1000*64*1280 f16 = 163,840,000 B
//   [163,840,000)  Gb  : same
//   [327,680,000)  hbuf: 2 dir * 2 buf * 64*320 f16 = 163,840 B
//   [327,843,840)  barrier counters (256 B)
//   [327,844,096)  Xh  : 64000*320 f16 = 40,960,000 B
//   [368,804,096)  WtT : 2*1280*320 f16 = 1,638,400 B  (col-major W)
//   [370,442,496)  UtT : 2*1280*320 f16 = 1,638,400 B  (col-major U)
// ---------------------------------------------------------------------------

typedef __attribute__((ext_vector_type(16))) _Float16 v16h;
typedef __attribute__((ext_vector_type(8)))  float    v8f;

union Frag16 { v16h v; _Float16 h[16]; unsigned u[8]; };
union Acc8   { v8f  v; float    f[8]; };

#define TT 1000
#define BB 64
#define CC 320
#define UU 320
#define G4 1280
#define NWG_PER_DIR 20

// 16-byte async DMA: global -> LDS, tracked by ASYNCcnt.
__device__ __forceinline__ void async_copy16(unsigned lds_addr, const void* gptr)
{
    asm volatile("global_load_async_to_lds_b128 %0, %1, off"
                 :: "v"(lds_addr), "v"((unsigned long long)(size_t)gptr)
                 : "memory");
}
__device__ __forceinline__ void wait_async(int n)
{
    if (n == 0) asm volatile("s_wait_asynccnt 0x0" ::: "memory");
    else        asm volatile("s_wait_asynccnt 0x2" ::: "memory");
}
__device__ __forceinline__ unsigned lds_addr_of(const void* p)
{
    return (unsigned)(size_t)p;   // LDS aperture: low 32 bits = LDS byte offset
}

// ---------------------------------------------------------------------------
// Pass 0a: x fp32 -> f16 (vectorized 4-wide). 20,480,000 elems.
// ---------------------------------------------------------------------------
__global__ __launch_bounds__(256) void cvt_x_kernel(
    const float* __restrict__ x, _Float16* __restrict__ Xh)
{
    size_t i = ((size_t)blockIdx.x * 256 + threadIdx.x) * 4;
    float4 v = *(const float4*)(x + i);
    union { _Float16 h[4]; unsigned long long u; } o;
    o.h[0] = (_Float16)v.x; o.h[1] = (_Float16)v.y;
    o.h[2] = (_Float16)v.z; o.h[3] = (_Float16)v.w;
    *(unsigned long long*)(Xh + i) = o.u;
}

// ---------------------------------------------------------------------------
// Pass 0b: transpose W and U to f16 col-major [dir][col][k].
// grid = (320 k, 2 dir, 2 which), block = 256.
// ---------------------------------------------------------------------------
__global__ __launch_bounds__(256) void cvt_w_kernel(
    const float* __restrict__ Wf, const float* __restrict__ Wb,
    const float* __restrict__ Uf, const float* __restrict__ Ub,
    _Float16* __restrict__ WtT, _Float16* __restrict__ UtT)
{
    const int k = blockIdx.x, dir = blockIdx.y, which = blockIdx.z;
    const float* src = which ? (dir ? Ub : Uf) : (dir ? Wb : Wf);
    _Float16*    dst = (which ? UtT : WtT) + (size_t)dir * G4 * CC;
    for (int c = threadIdx.x; c < G4; c += 256)
        dst[(size_t)c * CC + k] = (_Float16)src[(size_t)k * G4 + c];
}

// ---------------------------------------------------------------------------
// Phase 1: gate pre-computation GEMM. grid = (1000 rg, 20 cg, 2 dir),
// block = 256 (8 waves). 64 rows x 64 cols per WG, K=320 in 10 panels of 32,
// double-buffered LDS filled by async b128 DMA (1 chunk/thread/tile/panel).
// ---------------------------------------------------------------------------
__global__ __launch_bounds__(256) void gates_kernel(
    const _Float16* __restrict__ Xh, const _Float16* __restrict__ WtT,
    const float* __restrict__ bf, const float* __restrict__ bb,
    _Float16* __restrict__ Gf, _Float16* __restrict__ Gb)
{
    __shared__ __align__(16) _Float16 Xs[2][64 * 40];  // rows x panel (pad 40)
    __shared__ __align__(16) _Float16 Ws[2][64 * 40];  // cols x panel (pad 40)

    const int rg = blockIdx.x, cg = blockIdx.y, dir = blockIdx.z;
    const float* bias = dir ? bb : bf;
    _Float16*    G    = dir ? Gb : Gf;
    const _Float16* Wg = WtT + (size_t)dir * G4 * CC;

    const int tid = threadIdx.x, lane = tid & 31, wave = tid >> 5;
    const int mi  = wave >> 1;
    const int ci0 = (wave & 1) * 2;

    // DMA assignment: thread -> (row 0..63, 16B chunk 0..3) within a panel.
    const int drow = tid >> 2, dchk = tid & 3;
    const unsigned lX0 = lds_addr_of(&Xs[0][drow * 40 + dchk * 8]);
    const unsigned lX1 = lds_addr_of(&Xs[1][drow * 40 + dchk * 8]);
    const unsigned lW0 = lds_addr_of(&Ws[0][drow * 40 + dchk * 8]);
    const unsigned lW1 = lds_addr_of(&Ws[1][drow * 40 + dchk * 8]);
    const _Float16* gX = Xh + (size_t)(rg * 64 + drow) * CC + dchk * 8;
    const _Float16* gW = Wg + (size_t)(cg * 64 + drow) * CC + dchk * 8;

    Acc8 acc0, acc1;
#pragma unroll
    for (int r = 0; r < 8; ++r) { acc0.f[r] = 0.f; acc1.f[r] = 0.f; }

    // Prologue: panel 0 in flight.
    async_copy16(lX0, gX);
    async_copy16(lW0, gW);

    for (int p = 0; p < 10; ++p) {
        const int buf = p & 1;
        if (p < 9) {                       // prefetch panel p+1
            async_copy16(buf ? lX0 : lX1, gX + (p + 1) * 32);
            async_copy16(buf ? lW0 : lW1, gW + (p + 1) * 32);
            wait_async(2);                 // panel p landed (loads retire in order)
        } else {
            wait_async(0);
        }
        __syncthreads();

        // A fragment (16x32): lane<16 -> M=lane, K {0..7}u{16..23} (pairs/VGPR).
        Frag16 a;
        {
            int m   = mi * 16 + (lane & 15);
            int hi8 = (lane >> 4) * 8;
            const _Float16* base = &Xs[buf][m * 40];
#pragma unroll
            for (int j = 0; j < 8; ++j) {
                int k = hi8 + ((j >= 4) ? 16 : 0) + 2 * (j & 3);
                a.u[j] = *(const unsigned*)(base + k);
            }
        }
#pragma unroll
        for (int t2 = 0; t2 < 2; ++t2) {
            Frag16 bfr;
            int c = (ci0 + t2) * 16 + (lane & 15);
            const _Float16* base = &Ws[buf][c * 40 + (lane >> 4) * 8];
#pragma unroll
            for (int j = 0; j < 8; ++j)
                bfr.u[j] = *(const unsigned*)(base + ((j >= 4) ? 16 : 0) + 2 * (j & 3));
            if (t2 == 0)
                acc0.v = __builtin_amdgcn_wmma_f32_16x16x32_f16(
                    false, a.v, false, bfr.v, (short)0, acc0.v, false, false);
            else
                acc1.v = __builtin_amdgcn_wmma_f32_16x16x32_f16(
                    false, a.v, false, bfr.v, (short)0, acc1.v, false, false);
        }
        __syncthreads();                   // done reading buf -> reusable
    }

    // Store G (f16) with bias; C/D layout: (M = r + 8*(lane>=16), N = lane&15).
    const int n = lane & 15, hi = lane >> 4;
#pragma unroll
    for (int t2 = 0; t2 < 2; ++t2) {
        const Acc8& A = t2 ? acc1 : acc0;
        int c   = (ci0 + t2) * 16 + n;
        int col = cg * 64 + c;
        float bv = bias[col];
#pragma unroll
        for (int r = 0; r < 8; ++r) {
            int row = rg * 64 + mi * 16 + r + 8 * hi;   // row = b*T + t
            int b = row / TT, t = row % TT;
            G[((size_t)t * BB + b) * G4 + col] = (_Float16)(A.f[r] + bv);
        }
    }
}

// ---------------------------------------------------------------------------
// Monotonic-counter grid barrier (per direction, 20 WGs, 1001 rounds).
// ---------------------------------------------------------------------------
__device__ __forceinline__ void grid_barrier(unsigned* cnt, unsigned target)
{
    __syncthreads();
    if (threadIdx.x == 0) {
        __threadfence();
        atomicAdd(cnt, 1u);
        while (__hip_atomic_load(cnt, __ATOMIC_RELAXED,
                                 __HIP_MEMORY_SCOPE_AGENT) < target) { }
        __threadfence();
    }
    __syncthreads();
}

// ---------------------------------------------------------------------------
// Phase 2: recurrent scan. grid = 40 blocks (20/dir), block = 256 (8 waves).
// ---------------------------------------------------------------------------
__global__ __launch_bounds__(256) void recur_kernel(
    const _Float16* __restrict__ Gf, const _Float16* __restrict__ Gb,
    const _Float16* __restrict__ UtT,
    const unsigned char* __restrict__ mask,
    _Float16* __restrict__ hbuf,     // [2 dir][2 buf][64*320] f16
    unsigned* __restrict__ bar,      // [2] monotonic counters (memset to 0)
    float* __restrict__ out)
{
    __shared__ __align__(16) _Float16 Ut[64 * 328];  // [c][k], pad 328 (16B rows)
    __shared__ float zbuf[64 * 68];
    __shared__ float cbuf[64 * 16];

    const int dir = blockIdx.x / NWG_PER_DIR;
    const int wg  = blockIdx.x % NWG_PER_DIR;
    const int u0  = wg * 16;

    const _Float16* G  = dir ? Gb : Gf;
    const _Float16* Ug = UtT + (size_t)dir * G4 * CC;   // col-major f16 U
    _Float16* hb  = hbuf + (size_t)dir * 2 * BB * UU;
    unsigned* cnt = bar + dir;
    float* seq = out;
    float* st  = out + (size_t)BB * TT * 640;

    const int tid = threadIdx.x, lane = tid & 31, wave = tid >> 5;
    const int mi = wave >> 1, ci0 = (wave & 1) * 2;

    // Async-DMA the transposed U slice into LDS once (resident for 1000 steps):
    // Ut[c][k] = U[k][gate*320 + u0 + ulocal], c = gate*16 + ulocal.
    for (int i = tid; i < 64 * 40; i += 256) {          // 2560 16B chunks
        int c = i / 40, chk = i % 40;
        int col = (c >> 4) * UU + u0 + (c & 15);
        async_copy16(lds_addr_of(&Ut[c * 328 + chk * 8]),
                     Ug + (size_t)col * CC + chk * 8);
    }
    for (int i = tid; i < 64 * 16; i += 256) cbuf[i] = 0.f;
    for (int i = tid; i < BB * 16; i += 256) {          // zero h buffer 0 slice
        int b = i >> 4, u = i & 15;
        hb[(size_t)b * UU + u0 + u] = (_Float16)0.f;
    }
    wait_async(0);
    grid_barrier(cnt, NWG_PER_DIR);                     // init barrier

    for (int s = 0; s < TT; ++s) {
        const int t = dir ? (TT - 1 - s) : s;
        const _Float16* hprev = hb + (size_t)(s & 1) * BB * UU;
        _Float16*       hnext = hb + (size_t)((s + 1) & 1) * BB * UU;

        // ---- z = h_prev @ U_slice : each wave owns 2 of the 16 tiles ----
        Acc8 acc0, acc1;
#pragma unroll
        for (int r = 0; r < 8; ++r) { acc0.f[r] = 0.f; acc1.f[r] = 0.f; }

        for (int kb = 0; kb < UU; kb += 32) {
            Frag16 a;
            {
                int m   = mi * 16 + (lane & 15);
                int hi8 = (lane >> 4) * 8;
                const _Float16* abase = hprev + (size_t)m * UU + kb;
#pragma unroll
                for (int j = 0; j < 8; ++j) {
                    int k = hi8 + ((j >= 4) ? 16 : 0) + 2 * (j & 3);
                    a.u[j] = *(const unsigned*)(abase + k);
                }
            }
#pragma unroll
            for (int t2 = 0; t2 < 2; ++t2) {
                Frag16 bfr;
                int c = (ci0 + t2) * 16 + (lane & 15);
                const _Float16* bbase = &Ut[c * 328 + kb + (lane >> 4) * 8];
#pragma unroll
                for (int j = 0; j < 8; ++j)
                    bfr.u[j] = *(const unsigned*)(bbase + ((j >= 4) ? 16 : 0) + 2 * (j & 3));
                if (t2 == 0)
                    acc0.v = __builtin_amdgcn_wmma_f32_16x16x32_f16(
                        false, a.v, false, bfr.v, (short)0, acc0.v, false, false);
                else
                    acc1.v = __builtin_amdgcn_wmma_f32_16x16x32_f16(
                        false, a.v, false, bfr.v, (short)0, acc1.v, false, false);
            }
        }
        // Spill z tiles to LDS for cross-gate elementwise stage.
        {
            int n = lane & 15, hi = lane >> 4;
#pragma unroll
            for (int t2 = 0; t2 < 2; ++t2) {
                const Acc8& A = t2 ? acc1 : acc0;
                int c = (ci0 + t2) * 16 + n;
#pragma unroll
                for (int r = 0; r < 8; ++r)
                    zbuf[(mi * 16 + r + 8 * hi) * 68 + c] = A.f[r];
            }
        }
        __syncthreads();

        // ---- gates + state update (1024 (b,u) pairs, 4 per thread) ----
        for (int e = tid; e < BB * 16; e += 256) {
            int b = e >> 4, u = e & 15;
            const _Float16* Grow = G + ((size_t)t * BB + b) * G4 + u0 + u;
            float zi = zbuf[b * 68 + 0 * 16 + u] + (float)Grow[0];
            float zf = zbuf[b * 68 + 1 * 16 + u] + (float)Grow[UU];
            float zg = zbuf[b * 68 + 2 * 16 + u] + (float)Grow[2 * UU];
            float zo = zbuf[b * 68 + 3 * 16 + u] + (float)Grow[3 * UU];
            float ig = 1.f / (1.f + __expf(-zi));
            float fg = 1.f / (1.f + __expf(-zf));
            float gg = tanhf(zg);
            float og = 1.f / (1.f + __expf(-zo));
            float cn = fg * cbuf[e] + ig * gg;
            float hn = og * tanhf(cn);
            float hout;
            if (mask[(size_t)b * TT + t] != 0) { cbuf[e] = cn; hout = hn; }
            else { hout = (float)hprev[(size_t)b * UU + u0 + u]; }  // carry
            hnext[(size_t)b * UU + u0 + u] = (_Float16)hout;
            seq[((size_t)b * TT + t) * 640 + dir * UU + u0 + u] = hout;
            if (s == TT - 1)
                st[(size_t)b * 640 + dir * UU + u0 + u] = hout;
        }
        grid_barrier(cnt, (unsigned)NWG_PER_DIR * (unsigned)(s + 2));
    }
}

// ---------------------------------------------------------------------------
extern "C" void kernel_launch(void* const* d_in, const int* in_sizes, int n_in,
                              void* d_out, int out_size, void* d_ws, size_t ws_size,
                              hipStream_t stream)
{
    (void)in_sizes; (void)n_in; (void)out_size; (void)ws_size;
    const float*         x    = (const float*)d_in[0];
    const unsigned char* mask = (const unsigned char*)d_in[1];   // jnp bool = 1 byte
    const float* Wf = (const float*)d_in[2];
    const float* Uf = (const float*)d_in[3];
    const float* bf = (const float*)d_in[4];
    const float* Wb = (const float*)d_in[5];
    const float* Ub = (const float*)d_in[6];
    const float* bb = (const float*)d_in[7];
    float* out = (float*)d_out;

    char* ws = (char*)d_ws;
    const size_t g_elems = (size_t)TT * BB * G4;
    _Float16* Gf   = (_Float16*)ws;
    _Float16* Gb   = Gf + g_elems;
    _Float16* hbuf = Gb + g_elems;
    char* p_bar = ws + 2 * g_elems * sizeof(_Float16)
                     + 2 * 2 * (size_t)BB * UU * sizeof(_Float16);
    unsigned* bar  = (unsigned*)p_bar;
    _Float16* Xh   = (_Float16*)(p_bar + 256);
    _Float16* WtT  = Xh  + (size_t)BB * TT * CC;
    _Float16* UtT  = WtT + (size_t)2 * G4 * CC;

    hipMemsetAsync(bar, 0, 256, stream);

    cvt_x_kernel<<<20000, 256, 0, stream>>>(x, Xh);
    cvt_w_kernel<<<dim3(CC, 2, 2), 256, 0, stream>>>(Wf, Wb, Uf, Ub, WtT, UtT);
    gates_kernel<<<dim3(1000, 20, 2), 256, 0, stream>>>(Xh, WtT, bf, bb, Gf, Gb);
    recur_kernel<<<2 * NWG_PER_DIR, 256, 0, stream>>>(Gf, Gb, UtT, mask,
                                                      hbuf, bar, out);
}
